// NTM_29566554866015
// MI455X (gfx1250) — compile-verified, matrix operand order checked by reading, not run on hardware
//
#include <hip/hip_runtime.h>
#include <hip/hip_bf16.h>
#include <math.h>

// ---------------- problem constants ----------------
#define BB   256
#define NN   2048
#define MM   128
#define VECL 256
#define HH   1024
#define EE   512
#define PP   134            // M + 1 + 1 + 3 + 1
#define KIN  384            // VEC + M
#define EPSF 1e-16f

typedef __bf16 v16bf __attribute__((ext_vector_type(16)));
typedef float  v8f   __attribute__((ext_vector_type(8)));
typedef float  f4v   __attribute__((ext_vector_type(4)));

__device__ __forceinline__ float sigmoidf_(float x) { return 1.0f / (1.0f + expf(-x)); }
__device__ __forceinline__ float softplusf_(float x) { return (x > 20.0f) ? x : log1pf(expf(x)); }

// Load one 16x32 bf16 WMMA fragment for this lane from a row-major f32 matrix.
// Layout (CDNA5 ISA 7.12.2, 16-bit A 16x32): lane&15 selects row; lane>>4
// selects K-half; elems 0..7 = K (kbase..kbase+7), elems 8..15 = K+16.
__device__ __forceinline__ v16bf load_frag_f32(const float* __restrict__ Mtx, int ld,
                                               int row, bool valid, int k0, int lane) {
  v16bf f;
  if (valid) {
    const float* p = Mtx + (size_t)row * ld + (k0 + ((lane >> 4) << 3));
#pragma unroll
    for (int e = 0; e < 8; ++e) f[e] = (__bf16)p[e];
#pragma unroll
    for (int e = 0; e < 8; ++e) f[e + 8] = (__bf16)p[e + 16];
  } else {
#pragma unroll
    for (int e = 0; e < 16; ++e) f[e] = (__bf16)0.0f;
  }
  return f;
}

// ---------------- register-blocked 64x64-per-wave dual-K GEMM ----------------
// C[Mr,Nc] = act( A0@B0^T (K0) + A1@B1^T (K1, optional) + bias0 (+bias1) )
// Requires Mr%64==0, Nc%64==0, K%32==0. 4x4 WMMA accumulators per wave:
// A fragments reused across 4 n-subtiles, B fragments across 4 m-subtiles,
// cutting redundant L2 operand traffic ~4x vs 1 tile/wave.
__global__ void __launch_bounds__(256)
wmma_gemm_dual64(const float* __restrict__ A0, const float* __restrict__ B0, int K0,
                 const float* __restrict__ A1, const float* __restrict__ B1, int K1,
                 const float* __restrict__ bias0, const float* __restrict__ bias1,
                 float* __restrict__ C, int Mr, int Nc, int act) {
  const int lane = threadIdx.x & 31;
  const int wid  = threadIdx.x >> 5;
  const int nstr = Nc >> 6;
  const int wave = blockIdx.x * (blockDim.x >> 5) + wid;
  const int ms = wave / nstr, ns = wave - ms * nstr;
  const int m0 = ms << 6, n0 = ns << 6;
  const int rsel = lane & 15;
  v8f acc[4][4];
#pragma unroll
  for (int i = 0; i < 4; ++i)
#pragma unroll
    for (int j = 0; j < 4; ++j) acc[i][j] = (v8f){};

  for (int k0 = 0; k0 < K0; k0 += 32) {
    v16bf a[4];
#pragma unroll
    for (int i = 0; i < 4; ++i)
      a[i] = load_frag_f32(A0, K0, m0 + 16 * i + rsel, true, k0, lane);
#pragma unroll
    for (int j = 0; j < 4; ++j) {
      v16bf b = load_frag_f32(B0, K0, n0 + 16 * j + rsel, true, k0, lane);
#pragma unroll
      for (int i = 0; i < 4; ++i)
        acc[i][j] = __builtin_amdgcn_wmma_f32_16x16x32_bf16(false, a[i], false, b,
                                                            (short)0, acc[i][j], false, false);
    }
  }
  if (K1 > 0) {
    for (int k0 = 0; k0 < K1; k0 += 32) {
      v16bf a[4];
#pragma unroll
      for (int i = 0; i < 4; ++i)
        a[i] = load_frag_f32(A1, K1, m0 + 16 * i + rsel, true, k0, lane);
#pragma unroll
      for (int j = 0; j < 4; ++j) {
        v16bf b = load_frag_f32(B1, K1, n0 + 16 * j + rsel, true, k0, lane);
#pragma unroll
        for (int i = 0; i < 4; ++i)
          acc[i][j] = __builtin_amdgcn_wmma_f32_16x16x32_bf16(false, a[i], false, b,
                                                              (short)0, acc[i][j], false, false);
      }
    }
  }
  const int halfsel = (lane >> 4) << 3;
#pragma unroll
  for (int j = 0; j < 4; ++j) {
    const int col = n0 + 16 * j + rsel;
    float bv = bias0 ? bias0[col] : 0.0f;
    if (bias1) bv += bias1[col];
#pragma unroll
    for (int i = 0; i < 4; ++i) {
      const int rbase = m0 + 16 * i + halfsel;
#pragma unroll
      for (int r = 0; r < 8; ++r) {
        float v = acc[i][j][r] + bv;
        if (act == 1) v = sigmoidf_(v);
        C[(size_t)(rbase + r) * Nc + col] = v;
      }
    }
  }
}

// ---------------- generic 16x16-per-wave GEMM for ragged small N ----------------
__global__ void wmma_gemm_bt(const float* __restrict__ A, const float* __restrict__ Bt,
                             const float* __restrict__ bias, float* __restrict__ C,
                             int Mr, int Nc, int K, int act) {
  const int lane = threadIdx.x & 31;
  const int wid  = threadIdx.x >> 5;
  const int tiles_n = (Nc + 15) >> 4;
  const int tiles_total = ((Mr + 15) >> 4) * tiles_n;
  const int tile = blockIdx.x * (blockDim.x >> 5) + wid;
  if (tile >= tiles_total) return;            // wave-uniform: EXEC stays all-ones
  const int mt = tile / tiles_n, nt = tile - mt * tiles_n;
  const int m0 = mt << 4, n0 = nt << 4;
  const int arow = m0 + (lane & 15);
  const int brow = n0 + (lane & 15);          // C column == Bt row
  const bool bvalid = brow < Nc;
  v8f acc = {};
  for (int k0 = 0; k0 < K; k0 += 32) {
    v16bf af = load_frag_f32(A, K, arow, true, k0, lane);
    v16bf bf = load_frag_f32(Bt, K, brow, bvalid, k0, lane);
    acc = __builtin_amdgcn_wmma_f32_16x16x32_bf16(false, af, false, bf, (short)0, acc, false, false);
  }
  const int col = n0 + (lane & 15);
  if (col < Nc) {
    const float bv = bias ? bias[col] : 0.0f;
    const int rbase = m0 + ((lane >> 4) << 3);
#pragma unroll
    for (int r = 0; r < 8; ++r) {
      float v = acc[r] + bv;
      if (act == 1) v = sigmoidf_(v);
      C[(size_t)(rbase + r) * Nc + col] = v;
    }
  }
}

// ---------------- LSTM elementwise: h = sig(o)*tanh(sig(f)*c_prev + sig(i)*tanh(g)) ----------------
__global__ void lstm_act(const float* __restrict__ gates, const float* __restrict__ c_prev,
                         float* __restrict__ h) {
  const int idx = blockIdx.x * blockDim.x + threadIdx.x;   // B*H = 262144
  const int b = idx >> 10, j = idx & 1023;
  const float* gr = gates + (size_t)b * (4 * HH);
  float iv = gr[j], fv = gr[HH + j], gv = gr[2 * HH + j], ov = gr[3 * HH + j];
  float c = sigmoidf_(fv) * c_prev[idx] + sigmoidf_(iv) * tanhf(gv);
  h[idx] = sigmoidf_(ov) * tanhf(c);
}

// ---------------- head parameter prep (one block of 128 threads per batch) ----------------
// scal layout per batch (8 floats): beta, g, s0, s1, s2, gamma, ||k||, 0
__global__ void head_params(const float* __restrict__ p_r, const float* __restrict__ wp,
                            float* __restrict__ k_r, float* __restrict__ k_w,
                            float* __restrict__ e_v, float* __restrict__ a_v,
                            float* __restrict__ scal_r, float* __restrict__ scal_w,
                            float* __restrict__ read0) {
  const int b = blockIdx.x, t = threadIdx.x;   // 128 threads
  __shared__ float sb[128];
  const float* pr = p_r + (size_t)b * PP;
  const float* pw = wp  + (size_t)b * (PP + 2 * MM);
  float kr = tanhf(pr[t]);
  float kw = tanhf(pw[t]);
  k_r[b * MM + t] = kr;
  k_w[b * MM + t] = kw;
  e_v[b * MM + t] = sigmoidf_(pw[PP + t]);
  a_v[b * MM + t] = tanhf(pw[PP + MM + t]);
  read0[b * MM + t] = 0.0f;                    // zero the read accumulator every call
  sb[t] = kr * kr; __syncthreads();
  for (int s = 64; s > 0; s >>= 1) { if (t < s) sb[t] += sb[t + s]; __syncthreads(); }
  float knr = sqrtf(sb[0]); __syncthreads();
  sb[t] = kw * kw; __syncthreads();
  for (int s = 64; s > 0; s >>= 1) { if (t < s) sb[t] += sb[t + s]; __syncthreads(); }
  float knw = sqrtf(sb[0]);
  if (t == 0) {
    float s0 = pr[130], s1 = pr[131], s2 = pr[132];
    float mx = fmaxf(s0, fmaxf(s1, s2));
    float e0 = expf(s0 - mx), e1 = expf(s1 - mx), e2 = expf(s2 - mx);
    float es = e0 + e1 + e2;
    scal_r[b * 8 + 0] = softplusf_(pr[128]);
    scal_r[b * 8 + 1] = sigmoidf_(pr[129]);
    scal_r[b * 8 + 2] = e0 / es; scal_r[b * 8 + 3] = e1 / es; scal_r[b * 8 + 4] = e2 / es;
    scal_r[b * 8 + 5] = 1.0f + softplusf_(pr[133]);
    scal_r[b * 8 + 6] = knr; scal_r[b * 8 + 7] = 0.0f;
    s0 = pw[130]; s1 = pw[131]; s2 = pw[132];
    mx = fmaxf(s0, fmaxf(s1, s2));
    e0 = expf(s0 - mx); e1 = expf(s1 - mx); e2 = expf(s2 - mx);
    es = e0 + e1 + e2;
    scal_w[b * 8 + 0] = softplusf_(pw[128]);
    scal_w[b * 8 + 1] = sigmoidf_(pw[129]);
    scal_w[b * 8 + 2] = e0 / es; scal_w[b * 8 + 3] = e1 / es; scal_w[b * 8 + 4] = e2 / es;
    scal_w[b * 8 + 5] = 1.0f + softplusf_(pw[133]);
    scal_w[b * 8 + 6] = knw; scal_w[b * 8 + 7] = 0.0f;
  }
}

// ---------------- fused content-sim pass: ONE read of memory for both heads + norms ----------------
// one wave per (b,n) row of 128 floats; lane loads a nontemporal float4 (stream, no L2 pollution).
__global__ void sim_kernel(const float* __restrict__ mem,
                           const float* __restrict__ k_r, const float* __restrict__ k_w,
                           const float* __restrict__ scal_r, const float* __restrict__ scal_w,
                           float* __restrict__ sim_r, float* __restrict__ sim_w) {
  const int lane = threadIdx.x & 31;
  const int gw = blockIdx.x * (blockDim.x >> 5) + (threadIdx.x >> 5);   // B*N waves
  const int b = gw >> 11;
  const size_t rowoff = (size_t)gw * MM + lane * 4;
  f4v mv  = __builtin_nontemporal_load(reinterpret_cast<const f4v*>(mem + rowoff));
  f4v kr4 = *reinterpret_cast<const f4v*>(k_r + b * MM + lane * 4);
  f4v kw4 = *reinterpret_cast<const f4v*>(k_w + b * MM + lane * 4);
  float dr  = mv[0] * kr4[0] + mv[1] * kr4[1] + mv[2] * kr4[2] + mv[3] * kr4[3];
  float dw  = mv[0] * kw4[0] + mv[1] * kw4[1] + mv[2] * kw4[2] + mv[3] * kw4[3];
  float nsq = mv[0] * mv[0] + mv[1] * mv[1] + mv[2] * mv[2] + mv[3] * mv[3];
#pragma unroll
  for (int off = 16; off > 0; off >>= 1) {
    dr  += __shfl_xor(dr, off, 32);
    dw  += __shfl_xor(dw, off, 32);
    nsq += __shfl_xor(nsq, off, 32);
  }
  if (lane == 0) {
    float nrm = sqrtf(nsq);
    sim_r[gw] = scal_r[b * 8 + 0] * (dr / (nrm * scal_r[b * 8 + 6] + EPSF));
    sim_w[gw] = scal_w[b * 8 + 0] * (dw / (nrm * scal_w[b * 8 + 6] + EPSF));
  }
}

// ---------------- addressing: softmax -> gate -> circular 3-tap conv -> sharpen ----------------
// one block of 256 threads per batch; w_g row (2048) staged in LDS for the shift conv.
__global__ void address_kernel(const float* __restrict__ sim, const float* __restrict__ prev_w,
                               const float* __restrict__ scal, float* __restrict__ w_out) {
  const int b = blockIdx.x, t = threadIdx.x;  // 256 threads, 8 elems each
  __shared__ float wg[NN];
  __shared__ float rbuf[256];
  const float* srow = sim + (size_t)b * NN;
  const float* pw   = prev_w + (size_t)b * NN;
  float v[8];
  float lmax = -3.4e38f;
#pragma unroll
  for (int i = 0; i < 8; ++i) { v[i] = srow[t + i * 256]; lmax = fmaxf(lmax, v[i]); }
  rbuf[t] = lmax; __syncthreads();
  for (int s = 128; s > 0; s >>= 1) { if (t < s) rbuf[t] = fmaxf(rbuf[t], rbuf[t + s]); __syncthreads(); }
  const float gmax = rbuf[0]; __syncthreads();
  float lsum = 0.0f;
#pragma unroll
  for (int i = 0; i < 8; ++i) { v[i] = expf(v[i] - gmax); lsum += v[i]; }
  rbuf[t] = lsum; __syncthreads();
  for (int s = 128; s > 0; s >>= 1) { if (t < s) rbuf[t] += rbuf[t + s]; __syncthreads(); }
  const float inv_sum = 1.0f / rbuf[0]; __syncthreads();
  const float g  = scal[b * 8 + 1];
  const float s0 = scal[b * 8 + 2], s1 = scal[b * 8 + 3], s2 = scal[b * 8 + 4];
  const float gamma = scal[b * 8 + 5];
#pragma unroll
  for (int i = 0; i < 8; ++i) {
    const int idx = t + i * 256;
    wg[idx] = g * (v[i] * inv_sum) + (1.0f - g) * pw[idx];
  }
  __syncthreads();
  float wt[8];
  float lsum2 = 0.0f;
#pragma unroll
  for (int i = 0; i < 8; ++i) {
    const int idx = t + i * 256;
    float wtv = s0 * wg[(idx + NN - 1) & (NN - 1)] + s1 * wg[idx] + s2 * wg[(idx + 1) & (NN - 1)];
    wtv = powf(wtv, gamma);
    wt[i] = wtv; lsum2 += wtv;
  }
  rbuf[t] = lsum2; __syncthreads();
  for (int s = 128; s > 0; s >>= 1) { if (t < s) rbuf[t] += rbuf[t + s]; __syncthreads(); }
  const float inv2 = 1.0f / (rbuf[0] + EPSF);
#pragma unroll
  for (int i = 0; i < 8; ++i) w_out[(size_t)b * NN + t + i * 256] = wt[i] * inv2;
}

// ---------------- fused memory update + weighted read (one NT read + one NT write of memory) ----------------
// block handles (b, 64 rows); 8 waves x 8 rows; lane owns 4 m-columns.
__global__ void update_read(const float* __restrict__ mem,
                            const float* __restrict__ w_r, const float* __restrict__ w_w,
                            const float* __restrict__ e_v, const float* __restrict__ a_v,
                            float* __restrict__ new_mem, float* __restrict__ read_acc) {
  const int b  = blockIdx.x >> 5;             // 32 chunks per batch
  const int n0 = (blockIdx.x & 31) * 64;
  const int lane = threadIdx.x & 31, wv = threadIdx.x >> 5;
  __shared__ float sred[MM];
  const int m = lane * 4;
  f4v e4 = *reinterpret_cast<const f4v*>(e_v + b * MM + m);
  f4v a4 = *reinterpret_cast<const f4v*>(a_v + b * MM + m);
  float r0 = 0.f, r1 = 0.f, r2 = 0.f, r3 = 0.f;
#pragma unroll
  for (int r = 0; r < 8; ++r) {
    const int n = n0 + wv * 8 + r;
    const float wr = w_r[(size_t)b * NN + n];
    const float ww = w_w[(size_t)b * NN + n];
    const size_t off = ((size_t)b * NN + n) * MM + m;
    f4v mv = __builtin_nontemporal_load(reinterpret_cast<const f4v*>(mem + off));
    f4v nm;
    nm[0] = mv[0] * (1.0f - ww * e4[0]) + ww * a4[0];
    nm[1] = mv[1] * (1.0f - ww * e4[1]) + ww * a4[1];
    nm[2] = mv[2] * (1.0f - ww * e4[2]) + ww * a4[2];
    nm[3] = mv[3] * (1.0f - ww * e4[3]) + ww * a4[3];
    __builtin_nontemporal_store(nm, reinterpret_cast<f4v*>(new_mem + off));
    r0 += wr * mv[0]; r1 += wr * mv[1]; r2 += wr * mv[2]; r3 += wr * mv[3];
  }
  if (threadIdx.x < MM) sred[threadIdx.x] = 0.0f;
  __syncthreads();
  atomicAdd(&sred[m + 0], r0);
  atomicAdd(&sred[m + 1], r1);
  atomicAdd(&sred[m + 2], r2);
  atomicAdd(&sred[m + 3], r3);
  __syncthreads();
  if (threadIdx.x < MM) atomicAdd(&read_acc[b * MM + threadIdx.x], sred[threadIdx.x]);
}

// ---------------- concat [h | read] -> cat [B, H+M] ----------------
__global__ void concat_kernel(const float* __restrict__ h, const float* __restrict__ readv,
                              float* __restrict__ cat) {
  const int idx = blockIdx.x * blockDim.x + threadIdx.x;   // B*(H+M) = 294912
  const int b = idx / (HH + MM), c = idx - b * (HH + MM);
  cat[idx] = (c < HH) ? h[b * HH + c] : readv[b * MM + (c - HH)];
}

// ---------------- launch ----------------
extern "C" void kernel_launch(void* const* d_in, const int* in_sizes, int n_in,
                              void* d_out, int out_size, void* d_ws, size_t ws_size,
                              hipStream_t stream) {
  const float* x      = (const float*)d_in[0];
  const float* mem    = (const float*)d_in[1];
  const float* prev_r = (const float*)d_in[2];
  const float* prev_w = (const float*)d_in[3];
  const float* h_prev = (const float*)d_in[4];
  const float* c_prev = (const float*)d_in[5];
  const float* W_ih   = (const float*)d_in[6];
  const float* W_hh   = (const float*)d_in[7];
  const float* b_ih   = (const float*)d_in[8];
  const float* b_hh   = (const float*)d_in[9];
  const float* W_r    = (const float*)d_in[10];
  const float* b_r    = (const float*)d_in[11];
  const float* W_w    = (const float*)d_in[12];
  const float* b_w    = (const float*)d_in[13];
  const float* W_fc   = (const float*)d_in[14];
  const float* b_fc   = (const float*)d_in[15];

  float* out     = (float*)d_out;                    // [256,512]
  float* new_mem = (float*)d_out + (size_t)BB * EE;  // [256,2048,128]

  // workspace layout (floats, all 16B aligned)
  float* ws     = (float*)d_ws;
  float* gates  = ws;                                // 256*4096
  float* h      = gates  + (size_t)BB * 4 * HH;      // 256*1024
  float* p_r    = h      + (size_t)BB * HH;          // 256*134
  float* wp     = p_r    + (size_t)BB * PP;          // 256*390
  float* k_r    = wp     + (size_t)BB * (PP + 2 * MM);
  float* k_w    = k_r    + (size_t)BB * MM;
  float* e_v    = k_w    + (size_t)BB * MM;
  float* a_v    = e_v    + (size_t)BB * MM;
  float* scal_r = a_v    + (size_t)BB * MM;          // 256*8
  float* scal_w = scal_r + (size_t)BB * 8;
  float* sim_r  = scal_w + (size_t)BB * 8;           // 256*2048
  float* sim_w  = sim_r  + (size_t)BB * NN;
  float* w_rd   = sim_w  + (size_t)BB * NN;
  float* w_wr   = w_rd   + (size_t)BB * NN;
  float* readv  = w_wr   + (size_t)BB * NN;          // 256*128
  float* cat    = readv  + (size_t)BB * MM;          // 256*1152

  // 1) LSTM gates GEMM: gates = x@Wih^T (K=384) + h_prev@Whh^T (K=1024) + bih + bhh
  //    waves = (256/64)*(4096/64) = 256 -> 32 blocks of 8 waves
  wmma_gemm_dual64<<<32, 256, 0, stream>>>(x, W_ih, KIN, h_prev, W_hh, HH,
                                           b_ih, b_hh, gates, BB, 4 * HH, 0);
  // 2) LSTM elementwise -> h
  lstm_act<<<(BB * HH) / 256, 256, 0, stream>>>(gates, c_prev, h);
  // 3) head GEMMs: p_r = h@W_r^T+b_r (N=134), wp = h@W_w^T+b_w (N=390)
  wmma_gemm_bt<<<18, 256, 0, stream>>>(h, W_r, b_r, p_r, BB, PP, HH, 0);
  wmma_gemm_bt<<<50, 256, 0, stream>>>(h, W_w, b_w, wp, BB, PP + 2 * MM, HH, 0);
  // 4) head param activations + scalars + zero read accumulator
  head_params<<<BB, 128, 0, stream>>>(p_r, wp, k_r, k_w, e_v, a_v, scal_r, scal_w, readv);
  // 5) fused content-sim (single memory read for both heads + norms)
  sim_kernel<<<(BB * NN) / 8, 256, 0, stream>>>(mem, k_r, k_w, scal_r, scal_w, sim_r, sim_w);
  // 6) addressing for each head
  address_kernel<<<BB, 256, 0, stream>>>(sim_r, prev_r, scal_r, w_rd);
  address_kernel<<<BB, 256, 0, stream>>>(sim_w, prev_w, scal_w, w_wr);
  // 7) fused erase/add update + weighted read (writes new_memory into d_out)
  update_read<<<BB * (NN / 64), 256, 0, stream>>>(mem, w_rd, w_wr, e_v, a_v, new_mem, readv);
  // 8) concat [h | read]
  concat_kernel<<<(BB * (HH + MM)) / 256, 256, 0, stream>>>(h, readv, cat);
  // 9) out = sigmoid(cat @ W_fc^T + b_fc), waves = 4*8 = 32 -> 4 blocks
  wmma_gemm_dual64<<<4, 256, 0, stream>>>(cat, W_fc, HH + MM, nullptr, nullptr, 0,
                                          b_fc, nullptr, out, BB, EE, 1);
}
